// Attention_76321568850409
// MI455X (gfx1250) — matrix operand order repaired and checked by reading, not compiled, not run on hardware
//
#include <hip/hip_runtime.h>

typedef __attribute__((ext_vector_type(16))) __bf16 v16bf;
typedef __attribute__((ext_vector_type(2)))  __bf16 bf2;
typedef __attribute__((ext_vector_type(8)))  float  v8f;
typedef __attribute__((ext_vector_type(4)))  unsigned int u32x4;
typedef __attribute__((ext_vector_type(8)))  int          i32x8;
typedef __attribute__((ext_vector_type(4)))  int          i32x4;

constexpr int C     = 192;
constexpr int B     = 8;
constexpr int HEADS = 8;
constexpr int CPH   = 24;     // channels per head
constexpr int HH    = 128, WW = 128;
constexpr int N     = HH * WW; // 16384 spatial

// ---------------------------------------------------------------------------
// TDM: 2D tile load global -> LDS. Builds the D# (groups 0/1 per CDNA5 ISA
// 8.3/8.4) in SGPRs and issues tensor_load_to_lds (6-arg builtin).
// data_size_code: 0=1B 1=2B 2=4B 3=8B. Dims/strides in elements.
// ---------------------------------------------------------------------------
__device__ __forceinline__ void tdm_load_2d(unsigned int lds_off, const void* gsrc,
                                            int data_size_code,
                                            unsigned int tensor_d0, unsigned int tensor_d1,
                                            unsigned int tile_d0, unsigned int tile_d1,
                                            unsigned long long d0_stride) {
  const unsigned long long ga = (unsigned long long)gsrc;
  u32x4 g0;
  g0[0] = (unsigned int)__builtin_amdgcn_readfirstlane(1u);          // count=1 (valid user D#)
  g0[1] = (unsigned int)__builtin_amdgcn_readfirstlane(lds_off);     // lds_addr
  g0[2] = (unsigned int)__builtin_amdgcn_readfirstlane((unsigned int)ga);
  g0[3] = (unsigned int)__builtin_amdgcn_readfirstlane(
      (unsigned int)((ga >> 32) & 0x1FFFFFFu) | (2u << 30));         // addr[56:32] | type=2
  i32x8 g1;
  g1[0] = __builtin_amdgcn_readfirstlane((int)((unsigned)data_size_code << 16)); // wg_mask=0
  g1[1] = __builtin_amdgcn_readfirstlane((int)((tensor_d0 & 0xFFFFu) << 16));    // dim0[15:0] @bit48
  g1[2] = __builtin_amdgcn_readfirstlane(
      (int)(((tensor_d0 >> 16) & 0xFFFFu) | ((tensor_d1 & 0xFFFFu) << 16)));     // dim0[31:16], dim1[15:0]
  g1[3] = __builtin_amdgcn_readfirstlane(
      (int)(((tensor_d1 >> 16) & 0xFFFFu) | ((tile_d0 & 0xFFFFu) << 16)));       // dim1[31:16], tile0
  g1[4] = __builtin_amdgcn_readfirstlane((int)(tile_d1 & 0xFFFFu));              // tile1, tile2=0
  g1[5] = __builtin_amdgcn_readfirstlane((int)(d0_stride & 0xFFFFFFFFull));      // stride0[31:0]
  g1[6] = __builtin_amdgcn_readfirstlane((int)((d0_stride >> 32) & 0xFFFFull));  // stride0[47:32]
  g1[7] = 0;
  i32x4 z4;
  z4[0] = 0; z4[1] = 0; z4[2] = 0; z4[3] = 0;
  i32x8 z8;
  z8[0] = 0; z8[1] = 0; z8[2] = 0; z8[3] = 0;
  z8[4] = 0; z8[5] = 0; z8[6] = 0; z8[7] = 0;
  __builtin_amdgcn_tensor_load_to_lds(g0, g1, z4, z4, z8, 0);
}

// ---------------------------------------------------------------------------
// GEMM: Y[b,m,n] = sum_k W[m,k] * X[b,k,n] + bias[m]
// Block = 8 waves, owns full M=192 x 32 N-columns. The K=192 x 32 X-slab is
// DMA'd into LDS once via TDM, then each wave computes 3 M-tiles x 16 cols
// with v_wmma_f32_16x16x32_bf16, reading B-fragments from LDS.
// ---------------------------------------------------------------------------
template<bool IN_BF16, bool OUT_F32, bool W_PER_BATCH>
__global__ __launch_bounds__(256)
void gemm192_wmma(const void* __restrict__ Xv, const __bf16* __restrict__ Wall,
                  const float* __restrict__ bias, void* __restrict__ Yv) {
  __shared__ float xtile_f32[C * 32];             // 24KB (f32) / reused as bf16 (12KB)
  __bf16* xtile_bf = (__bf16*)xtile_f32;

  const int lane  = threadIdx.x & 31;
  const int wave  = threadIdx.x >> 5;
  const int b     = blockIdx.z;
  const int n0blk = blockIdx.x * 32;

  // one TDM DMA per block: rows k=0..191, cols n0blk..n0blk+31
  if (wave == 0) {
    if (IN_BF16) {
      const __bf16* gsrc = (const __bf16*)Xv + (size_t)b * C * N + n0blk;
      tdm_load_2d((unsigned int)(size_t)xtile_bf, gsrc, 1, N, C, 32, C, N);
    } else {
      const float* gsrc = (const float*)Xv + (size_t)b * C * N + n0blk;
      tdm_load_2d((unsigned int)(size_t)xtile_f32, gsrc, 2, N, C, 32, C, N);
    }
    __builtin_amdgcn_s_wait_tensorcnt(0);
  }
  __syncthreads();

  const __bf16* W = Wall + (W_PER_BATCH ? (size_t)b * C * C : (size_t)0);

  // wave -> (n half, 3 M-tiles)
  const int nhalf  = wave & 1;
  const int mt_set = wave >> 1;                  // 0..3; tiles mt_set, mt_set+4, mt_set+8

  // A layout (16-bit A 16x32): lanes 0-15 -> M=lane, K halves {0-7,16-23};
  // lanes 16-31 -> M=lane-16, K halves {8-15,24-31}.
  const int mA  = lane & 15;
  const int khA = (lane >> 4) << 3;              // 0 or 8
  // B layout (16-bit B 32x16): lanes 0-15 -> N=lane, K {0-7,16-23}; lanes 16-31 -> K {8-15,24-31}.
  const int colB = nhalf * 16 + (lane & 15);
  const int kbB  = (lane >> 4) << 3;             // 0 or 8

  v8f acc[3];
#pragma unroll
  for (int j = 0; j < 3; ++j) {
    const int m0 = (mt_set + 4 * j) * 16;
#pragma unroll
    for (int r = 0; r < 8; ++r) acc[j][r] = bias[m0 + ((lane < 16) ? r : r + 8)];
  }

#pragma unroll
  for (int kc = 0; kc < 6; ++kc) {
    const int k0 = kc * 32;
    v16bf bm;
#pragma unroll
    for (int i = 0; i < 8; ++i) {
      const int kk = k0 + kbB + ((i < 4) ? (2 * i) : (16 + 2 * (i - 4)));
      if (IN_BF16) {
        bm[2 * i]     = xtile_bf[kk * 32 + colB];
        bm[2 * i + 1] = xtile_bf[(kk + 1) * 32 + colB];
      } else {
        bm[2 * i]     = (__bf16)xtile_f32[kk * 32 + colB];
        bm[2 * i + 1] = (__bf16)xtile_f32[(kk + 1) * 32 + colB];
      }
    }
#pragma unroll
    for (int j = 0; j < 3; ++j) {
      const int m0 = (mt_set + 4 * j) * 16;
      v16bf a;
#pragma unroll
      for (int i = 0; i < 8; ++i) {
        const int kb = ((i < 4) ? (2 * i) : (16 + 2 * (i - 4))) + khA;
        bf2 w2 = *(const bf2*)(W + (size_t)(m0 + mA) * C + (k0 + kb));
        a[2 * i]     = w2.x;
        a[2 * i + 1] = w2.y;
      }
      acc[j] = __builtin_amdgcn_wmma_f32_16x16x32_bf16(false, a, false, bm,
                                                       (short)0, acc[j], false, false);
    }
  }

  const size_t ybase = (size_t)b * C * N;
#pragma unroll
  for (int j = 0; j < 3; ++j) {
    const int m0 = (mt_set + 4 * j) * 16;
#pragma unroll
    for (int r = 0; r < 8; ++r) {
      const int m = m0 + ((lane < 16) ? r : r + 8);
      const size_t off = ybase + (size_t)m * N + n0blk + nhalf * 16 + (lane & 15);
      if (OUT_F32) ((float*)Yv)[off]  = acc[j][r];
      else         ((__bf16*)Yv)[off] = (__bf16)acc[j][r];
    }
  }
}

// ---------------------------------------------------------------------------
// Depthwise 3x3 (SAME) + per-(b,c)-row sum-of-squares partials (for L2 norm).
// ---------------------------------------------------------------------------
__global__ __launch_bounds__(128)
void dwconv3x3_norm(const __bf16* __restrict__ in, const float* __restrict__ w,
                    const float* __restrict__ bias, __bf16* __restrict__ out,
                    float* __restrict__ partial) {
  const int bc = blockIdx.x;
  const int c  = bc % C;
  const int y  = blockIdx.y;
  const int x  = threadIdx.x;
  const size_t base = (size_t)bc * N;

  float acc = bias[c];
#pragma unroll
  for (int dy = 0; dy < 3; ++dy) {
    const int yy = y + dy - 1;
    if (yy < 0 || yy >= HH) continue;
#pragma unroll
    for (int dx = 0; dx < 3; ++dx) {
      const int xx = x + dx - 1;
      if (xx < 0 || xx >= WW) continue;
      acc += (float)in[base + (size_t)yy * WW + xx] * w[c * 9 + dy * 3 + dx];
    }
  }
  out[base + (size_t)y * WW + x] = (__bf16)acc;

  __shared__ float red[128];
  red[x] = acc * acc;
  __syncthreads();
#pragma unroll
  for (int s = 64; s > 0; s >>= 1) {
    if (x < s) red[x] += red[x + s];
    __syncthreads();
  }
  if (x == 0) partial[(size_t)bc * HH + y] = red[0];
}

__global__ __launch_bounds__(128)
void rownorm_reduce(const float* __restrict__ partial, float* __restrict__ norms) {
  __shared__ float red[128];
  const int x = threadIdx.x;
  red[x] = partial[(size_t)blockIdx.x * HH + x];
  __syncthreads();
#pragma unroll
  for (int s = 64; s > 0; s >>= 1) {
    if (x < s) red[x] += red[x + s];
    __syncthreads();
  }
  if (x == 0) norms[blockIdx.x] = red[0];
}

__global__ void cvt_weights(const float* __restrict__ wq, const float* __restrict__ wk,
                            const float* __restrict__ wv, __bf16* __restrict__ out) {
  const int i = blockIdx.x * 256 + threadIdx.x;
  const int m = C * C;
  if (i >= 3 * m) return;
  const float* src = (i < m) ? wq : (i < 2 * m) ? wk : wv;
  out[i] = (__bf16)src[i % m];
}

// ---------------------------------------------------------------------------
// Gram + softmax: G[c,d] = sum_n S[c,n]*V[d,n], scaled by 1/(||S_c|| ||V_d||) * T,
// softmax over d (24). One block per (which, head, b); 8 waves split K=16384.
// ---------------------------------------------------------------------------
__global__ __launch_bounds__(256)
void gram_softmax(const __bf16* __restrict__ qb, const __bf16* __restrict__ kb,
                  const __bf16* __restrict__ vb,
                  const float* __restrict__ nq, const float* __restrict__ nk,
                  const float* __restrict__ nv, const float* __restrict__ temp,
                  float* __restrict__ attn1, float* __restrict__ attn2) {
  const int which = blockIdx.x;   // 0 -> q (attn1), 1 -> k (attn2)
  const int head  = blockIdx.y;
  const int b     = blockIdx.z;
  const int lane  = threadIdx.x & 31;
  const int wave  = threadIdx.x >> 5;

  const __bf16* S  = which ? kb : qb;
  const float*  nS = which ? nk : nq;
  float* outp = which ? attn2 : attn1;

  const size_t rowbase = ((size_t)b * C + head * CPH) * N;

  const int mA  = lane & 15;
  const int khA = (lane >> 4) << 3;
  const int dB  = lane & 15;
  const int kbB = (lane >> 4) << 3;

  v8f acc[2][2];
#pragma unroll
  for (int mt = 0; mt < 2; ++mt)
#pragma unroll
    for (int dt = 0; dt < 2; ++dt)
#pragma unroll
      for (int r = 0; r < 8; ++r) acc[mt][dt][r] = 0.f;

  const int kstart = wave * (N / 8);
  for (int k0 = kstart; k0 < kstart + N / 8; k0 += 32) {
    v16bf am[2], bm[2];
#pragma unroll
    for (int mt = 0; mt < 2; ++mt) {
      const int row = mt * 16 + mA;
      if (row < CPH) {
#pragma unroll
        for (int i = 0; i < 8; ++i) {
          const int kb2 = ((i < 4) ? (2 * i) : (16 + 2 * (i - 4))) + khA;
          bf2 p = *(const bf2*)(S + rowbase + (size_t)row * N + k0 + kb2);
          am[mt][2 * i] = p.x; am[mt][2 * i + 1] = p.y;
        }
      } else {
#pragma unroll
        for (int i = 0; i < 16; ++i) am[mt][i] = (__bf16)0.f;
      }
    }
#pragma unroll
    for (int dt = 0; dt < 2; ++dt) {
      const int col = dt * 16 + dB;
      if (col < CPH) {
#pragma unroll
        for (int i = 0; i < 8; ++i) {
          bf2 p = *(const bf2*)(vb + rowbase + (size_t)col * N + k0 + kbB + ((i < 4) ? (2 * i) : (16 + 2 * (i - 4))));
          bm[dt][2 * i] = p.x; bm[dt][2 * i + 1] = p.y;
        }
      } else {
#pragma unroll
        for (int i = 0; i < 16; ++i) bm[dt][i] = (__bf16)0.f;
      }
    }
#pragma unroll
    for (int mt = 0; mt < 2; ++mt)
#pragma unroll
      for (int dt = 0; dt < 2; ++dt)
        acc[mt][dt] = __builtin_amdgcn_wmma_f32_16x16x32_bf16(
            false, am[mt], false, bm[dt], (short)0, acc[mt][dt], false, false);
  }

  __shared__ float gAll[8][32 * 32];
  float* slab = gAll[wave];
#pragma unroll
  for (int mt = 0; mt < 2; ++mt)
#pragma unroll
    for (int dt = 0; dt < 2; ++dt)
#pragma unroll
      for (int r = 0; r < 8; ++r) {
        const int row = mt * 16 + ((lane < 16) ? r : r + 8);
        const int col = dt * 16 + (lane & 15);
        slab[row * 32 + col] = acc[mt][dt][r];
      }
  __syncthreads();
  for (int cell = threadIdx.x; cell < 1024; cell += 256) {
    float s = 0.f;
#pragma unroll
    for (int w = 0; w < 8; ++w) s += gAll[w][cell];
    gAll[0][cell] = s;
  }
  __syncthreads();

  if (threadIdx.x < CPH) {
    const int row = threadIdx.x;
    const float t  = temp[head];
    const float is = 1.f / fmaxf(sqrtf(nS[b * C + head * CPH + row]), 1e-12f);
    float vals[CPH];
    float mx = -3.4e38f;
    for (int d = 0; d < CPH; ++d) {
      const float iv = 1.f / fmaxf(sqrtf(nv[b * C + head * CPH + d]), 1e-12f);
      vals[d] = gAll[0][row * 32 + d] * is * iv * t;
      mx = fmaxf(mx, vals[d]);
    }
    float sum = 0.f;
    for (int d = 0; d < CPH; ++d) { vals[d] = __expf(vals[d] - mx); sum += vals[d]; }
    const float inv = 1.f / sum;
    float* dst = outp + (((size_t)(b * HEADS + head) * CPH) + row) * CPH;
    for (int d = 0; d < CPH; ++d) dst[d] = vals[d] * inv;
  }
}

// P = attn1 @ attn2, per (b,head), 24x24.
__global__ __launch_bounds__(576)
void attn_chain(const float* __restrict__ a1, const float* __restrict__ a2,
                float* __restrict__ P) {
  const int bh = blockIdx.x;
  const int c  = threadIdx.x / CPH;
  const int d  = threadIdx.x % CPH;
  if (c >= CPH) return;
  const float* A  = a1 + (size_t)bh * CPH * CPH;
  const float* Bm = a2 + (size_t)bh * CPH * CPH;
  float s = 0.f;
  for (int e = 0; e < CPH; ++e) s += A[c * CPH + e] * Bm[e * CPH + d];
  P[(size_t)bh * CPH * CPH + (size_t)c * CPH + d] = s;
}

// Wcomb[b][o][hd*24+d] = (sum_c wproj[o][hd*24+c] * P[b,hd][c][d]) / ||v_d||
__global__ __launch_bounds__(256)
void combine_wproj(const float* __restrict__ wproj, const float* __restrict__ P,
                   const float* __restrict__ nv, __bf16* __restrict__ Wcomb) {
  const int e = blockIdx.x * 256 + threadIdx.x;
  const int b = blockIdx.y;
  if (e >= C * C) return;
  const int o  = e / C;
  const int cc = e % C;
  const int hd = cc / CPH;
  const int d  = cc % CPH;
  const float* Ph = P + (size_t)(b * HEADS + hd) * CPH * CPH;
  float s = 0.f;
  for (int c2 = 0; c2 < CPH; ++c2)
    s += wproj[o * C + hd * CPH + c2] * Ph[c2 * CPH + d];
  s *= 1.f / fmaxf(sqrtf(nv[b * C + cc]), 1e-12f);
  Wcomb[(size_t)b * C * C + e] = (__bf16)s;
}

// ---------------------------------------------------------------------------
extern "C" void kernel_launch(void* const* d_in, const int* in_sizes, int n_in,
                              void* d_out, int out_size, void* d_ws, size_t ws_size,
                              hipStream_t stream) {
  (void)in_sizes; (void)n_in; (void)out_size; (void)ws_size;
  const float* x      = (const float*)d_in[0];
  const float* edge   = (const float*)d_in[1];
  const float* grad   = (const float*)d_in[2];
  const float* w_q    = (const float*)d_in[3];  const float* b_q  = (const float*)d_in[4];
  const float* w_k    = (const float*)d_in[5];  const float* b_k  = (const float*)d_in[6];
  const float* w_v    = (const float*)d_in[7];  const float* b_v  = (const float*)d_in[8];
  const float* dw_q   = (const float*)d_in[9];  const float* db_q = (const float*)d_in[10];
  const float* dw_k   = (const float*)d_in[11]; const float* db_k = (const float*)d_in[12];
  const float* dw_v   = (const float*)d_in[13]; const float* db_v = (const float*)d_in[14];
  const float* w_proj = (const float*)d_in[15]; const float* b_proj = (const float*)d_in[16];
  const float* temp   = (const float*)d_in[17];

  char* ws = (char*)d_ws;
  size_t off = 0;
  auto take = [&](size_t bytes) -> void* {
    void* p = ws + off;
    off = (off + bytes + 255) & ~(size_t)255;
    return p;
  };
  constexpr size_t TENS = (size_t)B * C * N;
  __bf16* wbf   = (__bf16*)take((size_t)3 * C * C * sizeof(__bf16));
  __bf16* tmp   = (__bf16*)take(TENS * sizeof(__bf16));
  __bf16* qb    = (__bf16*)take(TENS * sizeof(__bf16));
  __bf16* kbuf  = (__bf16*)take(TENS * sizeof(__bf16));
  __bf16* vbuf  = (__bf16*)take(TENS * sizeof(__bf16));
  float*  part  = (float*)take((size_t)B * C * HH * sizeof(float));
  float*  nq    = (float*)take((size_t)B * C * sizeof(float));
  float*  nk    = (float*)take((size_t)B * C * sizeof(float));
  float*  nv    = (float*)take((size_t)B * C * sizeof(float));
  float*  attn1 = (float*)take((size_t)B * HEADS * CPH * CPH * sizeof(float));
  float*  attn2 = (float*)take((size_t)B * HEADS * CPH * CPH * sizeof(float));
  float*  P     = (float*)take((size_t)B * HEADS * CPH * CPH * sizeof(float));
  __bf16* Wcomb = (__bf16*)take((size_t)B * C * C * sizeof(__bf16));

  cvt_weights<<<(3 * C * C + 255) / 256, 256, 0, stream>>>(w_q, w_k, w_v, wbf);

  const dim3 ggrid(N / 32, 1, B);   // each block: full M=192 x 32 cols, LDS-staged via TDM
  // q = dwconv(conv1x1(edge))
  gemm192_wmma<false, false, false><<<ggrid, 256, 0, stream>>>(edge, wbf, b_q, tmp);
  dwconv3x3_norm<<<dim3(B * C, HH), 128, 0, stream>>>(tmp, dw_q, db_q, qb, part);
  rownorm_reduce<<<B * C, 128, 0, stream>>>(part, nq);
  // k = dwconv(conv1x1(grad_in))
  gemm192_wmma<false, false, false><<<ggrid, 256, 0, stream>>>(grad, wbf + C * C, b_k, tmp);
  dwconv3x3_norm<<<dim3(B * C, HH), 128, 0, stream>>>(tmp, dw_k, db_k, kbuf, part);
  rownorm_reduce<<<B * C, 128, 0, stream>>>(part, nk);
  // v = dwconv(conv1x1(x))
  gemm192_wmma<false, false, false><<<ggrid, 256, 0, stream>>>(x, wbf + 2 * C * C, b_v, tmp);
  dwconv3x3_norm<<<dim3(B * C, HH), 128, 0, stream>>>(tmp, dw_v, db_v, vbuf, part);
  rownorm_reduce<<<B * C, 128, 0, stream>>>(part, nv);

  gram_softmax<<<dim3(2, HEADS, B), 256, 0, stream>>>(qb, kbuf, vbuf, nq, nk, nv,
                                                      temp, attn1, attn2);
  attn_chain<<<B * HEADS, 576, 0, stream>>>(attn1, attn2, P);
  combine_wproj<<<dim3((C * C + 255) / 256, B), 256, 0, stream>>>(w_proj, P, nv, Wcomb);
  // final: d_out = Wcomb[b] @ v + b_proj  (f32 output)
  gemm192_wmma<true, true, true><<<ggrid, 256, 0, stream>>>(vbuf, Wcomb, b_proj, d_out);
}
